// POS_Tagger_82102594831139
// MI455X (gfx1250) — compile-verified, matrix operand order checked by reading, not run on hardware
//
#include <hip/hip_runtime.h>
#include <hip/hip_bf16.h>

// ---------------- problem constants (from reference) ----------------
#define B_N   2048      // N examples / outer sequence length
#define HDIM  100       // H
#define HP    128       // H padded to K-multiple of 32
#define G4    400       // 4*H (gates), divisible by 80
#define DW    128       // word embedding dim
#define CIN   528       // DW + 2H + 2H
#define CINP  544       // CIN padded to multiple of 32
#define LC    16
#define LB    24
#define NB    5         // N-tiles per wave (16x80 strip)

typedef __attribute__((ext_vector_type(16))) _Float16 v16h;
typedef __attribute__((ext_vector_type(8)))  _Float16 v8h;
typedef __attribute__((ext_vector_type(8)))  float    v8f;

// Load 32 K-values of one row with the CDNA5 16-bit A/B swizzle:
// element i -> K = (i&7) + 16*(i>>3) + 8*g, g = lane>>4. Two 16B vector loads.
__device__ __forceinline__ v16h ld_rowK(const _Float16* __restrict__ base, int ld,
                                        int row, int k0, int g) {
  const _Float16* p = base + (size_t)row * ld + k0 + 8 * g;
  v8h lo = *(const v8h*)(p);
  v8h hi = *(const v8h*)(p + 16);
  return __builtin_shufflevector(lo, hi, 0,1,2,3,4,5,6,7,8,9,10,11,12,13,14,15);
}

// Z[M,N] = bias[N] + A1[M,K1]·B1[N,K1]^T + A2[M,K2]·B2[N,K2]^T   (f16 in, f32 out)
// Each wave computes a 16x80 strip (1 M-tile x 5 N-tiles): the A fragment is
// loaded once per 32-K chunk and reused across 5 WMMAs. 8 waves / block.
__global__ __launch_bounds__(256)
void wmma_gemm_dual(const _Float16* __restrict__ A1, int lda1,
                    const _Float16* __restrict__ B1, int ldb1, int K1,
                    const _Float16* __restrict__ A2, int lda2,
                    const _Float16* __restrict__ B2, int ldb2, int K2,
                    const float* __restrict__ bias,
                    float* __restrict__ Z, int M, int N, int ldz) {
  const int lane = threadIdx.x & 31;
  const int wave = threadIdx.x >> 5;
  const int ngrp  = N / (16 * NB);           // N-strip groups (5 for N=400)
  const int total = (M >> 4) * ngrp;
  const int tile = blockIdx.x * 8 + wave;
  if (tile >= total) return;                 // wave-uniform: EXEC stays all-ones
  const int mt = tile / ngrp, ng = tile % ngrp;
  const int mr = mt << 4, nc0 = ng * (16 * NB);
  const int g  = lane >> 4;
  const int ln = lane & 15;

  v8f acc[NB];
#pragma unroll
  for (int j = 0; j < NB; ++j) {
    const float cb = bias ? bias[nc0 + j * 16 + ln] : 0.0f;  // C layout: lane -> N
#pragma unroll
    for (int i = 0; i < 8; ++i) acc[j][i] = cb;
  }

  const int arow = mr + ln;                  // A: lane -> M row
  for (int k0 = 0; k0 < K1; k0 += 32) {
    v16h a = ld_rowK(A1, lda1, arow, k0, g);
#pragma unroll
    for (int j = 0; j < NB; ++j) {           // B = W^T: lane -> W row (N)
      v16h b = ld_rowK(B1, ldb1, nc0 + j * 16 + ln, k0, g);
      acc[j] = __builtin_amdgcn_wmma_f32_16x16x32_f16(false, a, false, b, (short)0,
                                                      acc[j], false, false);
    }
  }
  for (int k0 = 0; k0 < K2; k0 += 32) {
    v16h a = ld_rowK(A2, lda2, arow, k0, g);
#pragma unroll
    for (int j = 0; j < NB; ++j) {
      v16h b = ld_rowK(B2, ldb2, nc0 + j * 16 + ln, k0, g);
      acc[j] = __builtin_amdgcn_wmma_f32_16x16x32_f16(false, a, false, b, (short)0,
                                                      acc[j], false, false);
    }
  }
  // D layout: element i -> M = mr + i + 8*g, lane -> N
#pragma unroll
  for (int j = 0; j < NB; ++j) {
    float* zp = Z + (size_t)(mr + 8 * g) * ldz + nc0 + j * 16 + ln;
#pragma unroll
    for (int i = 0; i < 8; ++i) zp[(size_t)i * ldz] = acc[j][i];
  }
}

// ---------------- small helper kernels ----------------
__global__ void pad_f16_kernel(const float* __restrict__ src, _Float16* __restrict__ dst,
                               int rows, int sc, int dc) {
  int idx = blockIdx.x * blockDim.x + threadIdx.x;
  if (idx >= rows * dc) return;
  int r = idx / dc, c = idx % dc;
  dst[idx] = (c < sc) ? (_Float16)src[(size_t)r * sc + c] : (_Float16)0.0f;
}

__global__ void add_bias_kernel(const float* a, const float* b, float* o, int n) {
  int i = blockIdx.x * blockDim.x + threadIdx.x;
  if (i < n) o[i] = a[i] + b[i];
}

__global__ void zero_f32_kernel(float* p, int n) {
  int i = blockIdx.x * blockDim.x + threadIdx.x;
  if (i < n) p[i] = 0.0f;
}
__global__ void zero_f16_kernel(_Float16* p, int n) {
  int i = blockIdx.x * blockDim.x + threadIdx.x;
  if (i < n) p[i] = (_Float16)0.0f;
}

// X[t][b][c] = emb[ids[b][ rev? clip(lens[b]-1-t) : t ]][c], f16, K-padded to HP
__global__ void embed_seq_kernel(const int* __restrict__ ids, const int* __restrict__ lens,
                                 const float* __restrict__ emb, _Float16* __restrict__ X,
                                 int L, int rev) {
  int idx = blockIdx.x * blockDim.x + threadIdx.x;
  if (idx >= L * B_N * HP) return;
  int c = idx & (HP - 1);
  int b = (idx >> 7) & (B_N - 1);
  int t = idx >> 18;
  int tt = t;
  if (rev) {
    int v = lens[b] - 1 - t;
    tt = v < 0 ? 0 : (v > L - 1 ? L - 1 : v);
  }
  int tok = ids[b * L + tt];
  X[idx] = (c < HDIM) ? (_Float16)emb[(size_t)tok * HDIM + c] : (_Float16)0.0f;
}

// gates + masked state update; emits f16 hidden for next step's WMMA GEMM
__global__ void lstm_gates_kernel(const float* __restrict__ Z, float* __restrict__ Hs,
                                  float* __restrict__ Cs, _Float16* __restrict__ Hf,
                                  const int* __restrict__ lens, int t) {
  int idx = blockIdx.x * blockDim.x + threadIdx.x;
  if (idx >= B_N * HP) return;
  int j = idx & (HP - 1);
  if (j >= HDIM) return;           // padding lanes stay zero
  int b = idx >> 7;
  const float* z = Z + (size_t)b * G4;
  float zi = z[j], zf = z[HDIM + j], zg = z[2 * HDIM + j], zo = z[3 * HDIM + j];
  float ig = 1.0f / (1.0f + __expf(-zi));
  float fg = 1.0f / (1.0f + __expf(-zf));
  float gg = tanhf(zg);
  float og = 1.0f / (1.0f + __expf(-zo));
  float cn = fg * Cs[idx] + ig * gg;
  float hn = og * tanhf(cn);
  bool m = t < lens[b];
  float hN = m ? hn : Hs[idx];
  float cN = m ? cn : Cs[idx];
  Hs[idx] = hN;
  Cs[idx] = cN;
  Hf[idx] = (_Float16)hN;
}

__global__ void copy_to_concat_kernel(const float* __restrict__ Hs,
                                      _Float16* __restrict__ cc, int coff) {
  int idx = blockIdx.x * blockDim.x + threadIdx.x;
  if (idx >= B_N * HDIM) return;
  int b = idx / HDIM, j = idx % HDIM;
  cc[(size_t)b * CINP + coff + j] = (_Float16)Hs[(size_t)b * HP + j];
}

__global__ void embed_word_kernel(const int* __restrict__ tokens,
                                  const float* __restrict__ emb,
                                  _Float16* __restrict__ cc) {
  int idx = blockIdx.x * blockDim.x + threadIdx.x;
  if (idx >= B_N * DW) return;
  int b = idx >> 7, c = idx & 127;
  cc[(size_t)b * CINP + c] = (_Float16)emb[(size_t)tokens[b] * DW + c];
}

// Persistent serial outer LSTM: block 0 = forward, block 1 = backward.
// Whh (400x100 f32 = 160KB) lives in LDS (CDNA5 WGP has 320KB).
__global__ void outer_lstm_seq_kernel(const float* __restrict__ Zx_f,
                                      const float* __restrict__ Zx_b,
                                      const float* __restrict__ Whh_f,
                                      const float* __restrict__ Whh_b,
                                      const float* __restrict__ bias_f,
                                      const float* __restrict__ bias_b,
                                      float* __restrict__ hs_f, float* __restrict__ hs_b,
                                      int T) {
  extern __shared__ float smem[];
  float* sW = smem;                 // 400*100
  float* sz = sW + G4 * HDIM;       // 400
  float* sh = sz + G4;              // 128
  float* sc = sh + HP;              // 128
  const int dir = blockIdx.x;
  const float* Zx = dir ? Zx_b : Zx_f;
  const float* W  = dir ? Whh_b : Whh_f;
  const float* bs = dir ? bias_b : bias_f;
  float* hs = dir ? hs_b : hs_f;
  const int tid = threadIdx.x;
  for (int i = tid; i < G4 * HDIM; i += blockDim.x) sW[i] = W[i];
  if (tid < HP) { sh[tid] = 0.0f; sc[tid] = 0.0f; }
  __syncthreads();
  for (int s = 0; s < T; ++s) {
    int t = dir ? (T - 1 - s) : s;
    if (tid < G4) {
      const float* wr = sW + tid * HDIM;
      float a = Zx[(size_t)t * G4 + tid] + bs[tid];
#pragma unroll 4
      for (int k = 0; k < HDIM; ++k) a += wr[k] * sh[k];
      sz[tid] = a;
    }
    __syncthreads();
    if (tid < HDIM) {
      float zi = sz[tid], zf = sz[HDIM + tid], zg = sz[2 * HDIM + tid], zo = sz[3 * HDIM + tid];
      float ig = 1.0f / (1.0f + __expf(-zi));
      float fg = 1.0f / (1.0f + __expf(-zf));
      float gg = tanhf(zg);
      float og = 1.0f / (1.0f + __expf(-zo));
      float cn = fg * sc[tid] + ig * gg;
      float hn = og * tanhf(cn);
      sc[tid] = cn; sh[tid] = hn;
      hs[(size_t)t * HDIM + tid] = hn;
    }
    __syncthreads();
  }
}

__global__ void proj_kernel(const float* __restrict__ hs_f, const float* __restrict__ hs_b,
                            const float* __restrict__ W_pos, const float* __restrict__ b_pos,
                            const float* __restrict__ W_freq, const float* __restrict__ b_freq,
                            float* __restrict__ out) {
  int idx = blockIdx.x * blockDim.x + threadIdx.x;
  if (idx >= B_N * 37) return;
  int b = idx / 37, j = idx % 37;
  const float* W;
  float acc;
  if (j < 17) { W = W_pos + j * (2 * HDIM); acc = b_pos[j]; }
  else        { W = W_freq + (j - 17) * (2 * HDIM); acc = b_freq[j - 17]; }
  const float* hf = hs_f + (size_t)b * HDIM;
  const float* hb = hs_b + (size_t)b * HDIM;
  for (int k = 0; k < HDIM; ++k) acc += W[k] * hf[k];
  for (int k = 0; k < HDIM; ++k) acc += W[HDIM + k] * hb[k];
  if (j < 17) out[(size_t)b * 17 + j] = acc;
  else        out[(size_t)B_N * 17 + (size_t)b * 20 + (j - 17)] = acc;
}

// ---------------- host orchestration ----------------
extern "C" void kernel_launch(void* const* d_in, const int* in_sizes, int n_in,
                              void* d_out, int out_size, void* d_ws, size_t ws_size,
                              hipStream_t stream) {
  (void)in_sizes; (void)n_in; (void)out_size; (void)ws_size;
  // pytree flatten order of setup_inputs():
  const int*   tokens    = (const int*)d_in[0];
  const int*   chars     = (const int*)d_in[1];
  const int*   char_lens = (const int*)d_in[2];
  const int*   byts      = (const int*)d_in[3];
  const int*   byte_lens = (const int*)d_in[4];
  const float* emb_word  = (const float*)d_in[5];
  const float* emb_char  = (const float*)d_in[6];
  const float* emb_byte  = (const float*)d_in[7];
  // char_params: 8=Wih_f 9=Whh_f 10=bih_f 11=bhh_f 12=Wih_b 13=Whh_b 14=bih_b 15=bhh_b
  // byte_params: 16..23 ; outer_params: 24..31
  const float* W_pos  = (const float*)d_in[32];
  const float* b_pos  = (const float*)d_in[33];
  const float* W_freq = (const float*)d_in[34];
  const float* b_freq = (const float*)d_in[35];

  char* ws = (char*)d_ws;
  size_t off = 0;
  auto alloc = [&](size_t bytes) -> void* {
    void* p = ws + off;
    off += (bytes + 255) & ~(size_t)255;
    return p;
  };
  _Float16* concatH = (_Float16*)alloc((size_t)B_N * CINP * 2);
  _Float16* Wih16[6];
  _Float16* Whh16[4];
  for (int i = 0; i < 4; ++i) Wih16[i] = (_Float16*)alloc((size_t)G4 * HP * 2);
  Wih16[4] = (_Float16*)alloc((size_t)G4 * CINP * 2);
  Wih16[5] = (_Float16*)alloc((size_t)G4 * CINP * 2);
  for (int i = 0; i < 4; ++i) Whh16[i] = (_Float16*)alloc((size_t)G4 * HP * 2);
  float*    bias6 = (float*)alloc((size_t)6 * G4 * 4);
  _Float16* Xseq  = (_Float16*)alloc((size_t)LB * B_N * HP * 2);
  float*    Zbuf  = (float*)alloc((size_t)B_N * G4 * 4);
  float*    Hs    = (float*)alloc((size_t)B_N * HP * 4);
  float*    Cs    = (float*)alloc((size_t)B_N * HP * 4);
  _Float16* Hf    = (_Float16*)alloc((size_t)B_N * HP * 2);
  float*    ZxOf  = (float*)alloc((size_t)B_N * G4 * 4);
  float*    ZxOb  = (float*)alloc((size_t)B_N * G4 * 4);
  float*    hsF   = (float*)alloc((size_t)B_N * HDIM * 4);
  float*    hsB   = (float*)alloc((size_t)B_N * HDIM * 4);

  auto gsz = [](int n, int b) { return (n + b - 1) / b; };

  // ---- weight prep: pad+convert to f16, fold bih+bhh ----
  struct WJob { int idx; _Float16* dst; int sc, dc; };
  const WJob wjobs[10] = {
    {8,  Wih16[0], HDIM, HP}, {12, Wih16[1], HDIM, HP},
    {16, Wih16[2], HDIM, HP}, {20, Wih16[3], HDIM, HP},
    {24, Wih16[4], CIN, CINP}, {28, Wih16[5], CIN, CINP},
    {9,  Whh16[0], HDIM, HP}, {13, Whh16[1], HDIM, HP},
    {17, Whh16[2], HDIM, HP}, {21, Whh16[3], HDIM, HP},
  };
  for (int i = 0; i < 10; ++i)
    pad_f16_kernel<<<gsz(G4 * wjobs[i].dc, 256), 256, 0, stream>>>(
        (const float*)d_in[wjobs[i].idx], wjobs[i].dst, G4, wjobs[i].sc, wjobs[i].dc);
  const int bidx[6][2] = {{10,11},{14,15},{18,19},{22,23},{26,27},{30,31}};
  for (int i = 0; i < 6; ++i)
    add_bias_kernel<<<gsz(G4, 256), 256, 0, stream>>>(
        (const float*)d_in[bidx[i][0]], (const float*)d_in[bidx[i][1]], bias6 + i * G4, G4);

  zero_f16_kernel<<<gsz(B_N * CINP, 256), 256, 0, stream>>>(concatH, B_N * CINP);
  embed_word_kernel<<<gsz(B_N * DW, 256), 256, 0, stream>>>(tokens, emb_word, concatH);

  const int waves = (B_N / 16) * (G4 / (16 * NB));   // 128 * 5 = 640 wave-strips

  // ---- char/byte BiLSTM encoders (WMMA per step: Z = bias + X_t·Wih^T + H·Whh^T) ----
  auto run_enc = [&](const int* ids, const int* lens, const float* emb,
                     _Float16* Wih, _Float16* Whh, const float* bias,
                     int L, int rev, int coff) {
    embed_seq_kernel<<<gsz(L * B_N * HP, 256), 256, 0, stream>>>(ids, lens, emb, Xseq, L, rev);
    zero_f32_kernel<<<gsz(B_N * HP, 256), 256, 0, stream>>>(Hs, B_N * HP);
    zero_f32_kernel<<<gsz(B_N * HP, 256), 256, 0, stream>>>(Cs, B_N * HP);
    zero_f16_kernel<<<gsz(B_N * HP, 256), 256, 0, stream>>>(Hf, B_N * HP);
    for (int t = 0; t < L; ++t) {
      wmma_gemm_dual<<<gsz(waves, 8), 256, 0, stream>>>(
          Xseq + (size_t)t * B_N * HP, HP, Wih, HP, HP,
          Hf, HP, Whh, HP, HP,
          bias, Zbuf, B_N, G4, G4);
      lstm_gates_kernel<<<gsz(B_N * HP, 256), 256, 0, stream>>>(Zbuf, Hs, Cs, Hf, lens, t);
    }
    copy_to_concat_kernel<<<gsz(B_N * HDIM, 256), 256, 0, stream>>>(Hs, concatH, coff);
  };
  run_enc(chars, char_lens, emb_char, Wih16[0], Whh16[0], bias6 + 0 * G4, LC, 0, DW);
  run_enc(chars, char_lens, emb_char, Wih16[1], Whh16[1], bias6 + 1 * G4, LC, 1, DW + HDIM);
  run_enc(byts,  byte_lens, emb_byte, Wih16[2], Whh16[2], bias6 + 2 * G4, LB, 0, DW + 2 * HDIM);
  run_enc(byts,  byte_lens, emb_byte, Wih16[3], Whh16[3], bias6 + 3 * G4, LB, 1, DW + 3 * HDIM);

  // ---- outer BiLSTM: precompute input contribution with WMMA, then serial scan ----
  wmma_gemm_dual<<<gsz(waves, 8), 256, 0, stream>>>(
      concatH, CINP, Wih16[4], CINP, CINP,
      (const _Float16*)nullptr, 0, (const _Float16*)nullptr, 0, 0,
      (const float*)nullptr, ZxOf, B_N, G4, G4);
  wmma_gemm_dual<<<gsz(waves, 8), 256, 0, stream>>>(
      concatH, CINP, Wih16[5], CINP, CINP,
      (const _Float16*)nullptr, 0, (const _Float16*)nullptr, 0, 0,
      (const float*)nullptr, ZxOb, B_N, G4, G4);
  const size_t smem = (size_t)(G4 * HDIM + G4 + HP + HP) * 4;   // ~163KB < 320KB LDS/WGP
  outer_lstm_seq_kernel<<<2, 512, smem, stream>>>(
      ZxOf, ZxOb, (const float*)d_in[25], (const float*)d_in[29],
      bias6 + 4 * G4, bias6 + 5 * G4, hsF, hsB, B_N);

  // ---- output projections ----
  proj_kernel<<<gsz(B_N * 37, 256), 256, 0, stream>>>(
      hsF, hsB, W_pos, b_pos, W_freq, b_freq, (float*)d_out);
}
